// PHPGenerator_55379308315087
// MI455X (gfx1250) — matrix operand order repaired and checked by reading, not compiled, not run on hardware
//
#include <hip/hip_runtime.h>
#include <hip/hip_bf16.h>
#include <hip/hip_fp16.h>

// ---------------------------------------------------------------------------
// Seq2seq LSTM encoder/decoder with attention for MI455X (gfx1250, wave32).
// Matmuls use v_wmma_f32_16x16x32_f16; the final vocab GEMM stages weight
// panels into LDS with CDNA5 async-to-LDS loads (double buffered).
// ---------------------------------------------------------------------------

typedef __attribute__((ext_vector_type(16))) _Float16 v16h;
typedef __attribute__((ext_vector_type(8)))  _Float16 v8h;
typedef __attribute__((ext_vector_type(8)))  float    v8f;

#define B_SZ   32
#define S_SRC  512
#define T_TGT  128
#define VOCAB  32000
#define EDIM   256
#define HDIM   512
#define G4H    (4 * HDIM)          // 2048
#define KENC   (EDIM + HDIM)       // 768  = [emb | h]
#define KDEC   (EDIM + 2 * HDIM)   // 1280 = [emb | ctx | h]

#define LDS_ROW 40                 // 32 halfs + 8 pad (80B row, bank-spread)

// ---------------- WMMA fragment loaders (row-major f16 sources) ------------
// A-matrix 16x32 (MxK): lanes 0-15 hold M=0..15 with K in kb..kb+15,
// kb = (lane>=16)?16:0.  Two contiguous 16B chunks per lane.
__device__ inline v16h load_afrag(const _Float16* __restrict__ A, int ld,
                                  int row0, int k0) {
  int lane = threadIdx.x & 31;
  int r  = lane & 15;
  int kb = (lane >> 4) << 4;
  const _Float16* p = A + (size_t)(row0 + r) * ld + k0 + kb;
  v8h lo = *(const v8h*)(p);
  v8h hi = *(const v8h*)(p + 8);
  v16h v;
#pragma unroll
  for (int i = 0; i < 8; ++i) { v[i] = lo[i]; v[8 + i] = hi[i]; }
  return v;
}

// B-matrix 32x16 (KxN) where B[k][n] = W[n][k] (W row-major [N,K], ld=K):
// lane n = lane&15, K window = (lane>=16)?16..31:0..15 -> one contiguous 32B load.
__device__ inline v16h load_bfrag(const _Float16* __restrict__ W, int ld,
                                  int n0, int k0) {
  int lane = threadIdx.x & 31;
  int n  = lane & 15;
  int kb = (lane >> 4) << 4;
  const _Float16* p = W + (size_t)(n0 + n) * ld + k0 + kb;
  return *(const v16h*)(p);
}

// Same B fragment but from an LDS panel [64][LDS_ROW] f16 (k-window = 32).
__device__ inline v16h load_bfrag_lds(const _Float16* sB, int nrow0) {
  int lane = threadIdx.x & 31;
  int n  = lane & 15;
  int kb = (lane >> 4) << 4;
  const _Float16* p = sB + (nrow0 + n) * LDS_ROW + kb;
  v8h lo = *(const v8h*)(p);
  v8h hi = *(const v8h*)(p + 8);
  v16h v;
#pragma unroll
  for (int i = 0; i < 8; ++i) { v[i] = lo[i]; v[8 + i] = hi[i]; }
  return v;
}

__device__ inline float sigf(float x) { return 1.0f / (1.0f + expf(-x)); }

// CDNA5 async global->LDS copy, 16B per lane (tracked by ASYNCcnt).
__device__ inline void async_copy_b128(unsigned lds_off, const void* gptr) {
  unsigned long long ga = (unsigned long long)(uintptr_t)gptr;
  asm volatile("global_load_async_to_lds_b128 %0, %1, off"
               : : "v"(lds_off), "v"(ga) : "memory");
}
__device__ inline void wait_asynccnt0() {
#if __has_builtin(__builtin_amdgcn_s_wait_asynccnt)
  __builtin_amdgcn_s_wait_asynccnt(0);
#else
  asm volatile("s_wait_asynccnt 0x0" ::: "memory");
#endif
}

// ---------------- small helper kernels -------------------------------------

// dst[r][0:w0] = (f16)src0[r][:], dst[r][w0:w0+w1] = (f16)src1[r][:]
__global__ void convert_concat_f16(const float* __restrict__ src0, int w0,
                                   const float* __restrict__ src1, int w1,
                                   _Float16* __restrict__ dst, long total) {
  long i = (long)blockIdx.x * blockDim.x + threadIdx.x;
  if (i >= total) return;
  int wt = w0 + w1;
  long r = i / wt;
  int  cc = (int)(i - r * wt);
  float v = (cc < w0) ? src0[r * (long)w0 + cc] : src1[r * (long)w1 + (cc - w0)];
  dst[i] = (_Float16)v;
}

// gather token embeddings -> f16 [ntok, EDIM]
__global__ void gather_embed_f16(const int* __restrict__ tokens,
                                 const float* __restrict__ table,
                                 _Float16* __restrict__ dst, long total) {
  long i = (long)blockIdx.x * blockDim.x + threadIdx.x;
  if (i >= total) return;
  long tok = i >> 8;            // /EDIM
  int  e   = (int)(i & 255);
  int  id  = tokens[tok];
  dst[i] = (_Float16)table[(size_t)id * EDIM + e];
}

__global__ void zero_state(float* __restrict__ h, float* __restrict__ c,
                           _Float16* __restrict__ h16) {
  int i = blockIdx.x * blockDim.x + threadIdx.x;  // 16384 = B*H
  h[i] = 0.f; c[i] = 0.f; h16[i] = (_Float16)0.f;
}

// xh_enc[b, 0:256] = enc_emb[b,t,:]; xh_enc[b, 256:768] = h16[b,:]
__global__ void pack_enc_xh(const _Float16* __restrict__ encEmb,
                            const _Float16* __restrict__ h16,
                            _Float16* __restrict__ xh, int t) {
  int i = blockIdx.x * blockDim.x + threadIdx.x;  // < 32*768
  int b = i / KENC, k = i - b * KENC;
  xh[i] = (k < EDIM) ? encEmb[((size_t)b * S_SRC + t) * EDIM + k]
                     : h16[b * HDIM + (k - EDIM)];
}

// ---------------- fused LSTM step: gates GEMM (WMMA) + pointwise ------------
// gates[b, g*H + j] = sum_k xh[b,k] * Wcat[g*H+j, k] + bias[g*H+j]
// Each wave owns (m-tile of 16 batch rows) x (16 hidden units j) across all
// 4 gates, so the LSTM cell update is wave-local.  Grid = 64 waves exactly.
__global__ void __launch_bounds__(128)
lstm_step_fused(const _Float16* __restrict__ xh, int K,
                const _Float16* __restrict__ Wcat,
                const float* __restrict__ bias,
                float* __restrict__ h, float* __restrict__ c,
                _Float16* __restrict__ h16,
                float* __restrict__ seq_f32,      // encoder: enc_out[b,t,:]
                _Float16* __restrict__ seq_f16,   // decoder: dec_hs[b,t,:]
                int t, int seqlen) {
  int wid  = (blockIdx.x * blockDim.x + threadIdx.x) >> 5;  // 0..63
  int lane = threadIdx.x & 31;
  int m0 = (wid & 1) << 4;          // batch tile base (0 or 16)
  int n0 = (wid >> 1) << 4;         // hidden unit tile base (0..496)
  int n  = lane & 15;
  int j  = n0 + n;                  // hidden index owned by this lane

  v8f acc[4];
#pragma unroll
  for (int g = 0; g < 4; ++g) {
    float bv = bias[g * HDIM + j];
#pragma unroll
    for (int i = 0; i < 8; ++i) acc[g][i] = bv;
  }

  for (int k0 = 0; k0 < K; k0 += 32) {
    v16h a = load_afrag(xh, K, m0, k0);
#pragma unroll
    for (int g = 0; g < 4; ++g) {
      v16h bm = load_bfrag(Wcat + (size_t)g * HDIM * K, K, n0, k0);
      acc[g] = __builtin_amdgcn_wmma_f32_16x16x32_f16(
          false, a, false, bm, (short)0, acc[g], false, false);
    }
  }

  // pointwise LSTM cell: VGPR r of C/D holds batch row m0 + r (+8 for hi lanes)
  int rbase = m0 + ((lane >> 4) << 3);
#pragma unroll
  for (int r = 0; r < 8; ++r) {
    int bb = rbase + r;
    float iv = sigf(acc[0][r]);
    float fv = sigf(acc[1][r]);
    float gv = tanhf(acc[2][r]);
    float ov = sigf(acc[3][r]);
    size_t off = (size_t)bb * HDIM + j;
    float cn = fv * c[off] + iv * gv;
    float hn = ov * tanhf(cn);
    c[off] = cn;
    h[off] = hn;
    h16[off] = (_Float16)hn;
    if (seq_f32) seq_f32[((size_t)bb * seqlen + t) * HDIM + j] = hn;
    if (seq_f16) seq_f16[((size_t)bb * seqlen + t) * HDIM + j] = (_Float16)hn;
  }
}

// ---------------- decoder attention ----------------------------------------
// scores[b,s] = dot(enc_out[b,s,:], h[b,:]); one wave per (b,s).
__global__ void attn_scores(const float* __restrict__ encOut,
                            const float* __restrict__ h,
                            float* __restrict__ scores) {
  int gw   = (blockIdx.x * blockDim.x + threadIdx.x) >> 5;  // 0..16383
  int lane = threadIdx.x & 31;
  int b = gw >> 9, s = gw & 511;
  const float* e  = encOut + ((size_t)b * S_SRC + s) * HDIM;
  const float* hb = h + (size_t)b * HDIM;
  float acc = 0.f;
#pragma unroll
  for (int k = lane; k < HDIM; k += 32) acc += e[k] * hb[k];
#pragma unroll
  for (int off = 16; off > 0; off >>= 1) acc += __shfl_down(acc, off, 32);
  if (lane == 0) scores[b * S_SRC + s] = acc;
}

// softmax over S, ctx = sum_s w*enc_out, then pack xh_dec = [emb|ctx|h16].
__global__ void __launch_bounds__(512)
attn_softmax_ctx_pack(const float* __restrict__ scores,
                      const float* __restrict__ encOut,
                      const _Float16* __restrict__ decEmb,
                      const _Float16* __restrict__ h16,
                      _Float16* __restrict__ xh, int t) {
  __shared__ float red[S_SRC];
  __shared__ float wgt[S_SRC];
  int b = blockIdx.x, tid = threadIdx.x;
  float sc = scores[b * S_SRC + tid];
  red[tid] = sc; __syncthreads();
  for (int s = 256; s > 0; s >>= 1) {
    if (tid < s) red[tid] = fmaxf(red[tid], red[tid + s]);
    __syncthreads();
  }
  float mx = red[0]; __syncthreads();
  float ev = expf(sc - mx);
  red[tid] = ev; __syncthreads();
  for (int s = 256; s > 0; s >>= 1) {
    if (tid < s) red[tid] += red[tid + s];
    __syncthreads();
  }
  wgt[tid] = ev / red[0]; __syncthreads();

  // context for hidden column tid (coalesced over H)
  float acc = 0.f;
  const float* eb = encOut + (size_t)b * S_SRC * HDIM + tid;
  for (int s = 0; s < S_SRC; ++s) acc += wgt[s] * eb[(size_t)s * HDIM];

  _Float16* xb = xh + (size_t)b * KDEC;
  xb[EDIM + tid]        = (_Float16)acc;                              // ctx
  xb[EDIM + HDIM + tid] = h16[b * HDIM + tid];                        // h
  if (tid < EDIM)
    xb[tid] = decEmb[((size_t)b * T_TGT + t) * EDIM + tid];           // emb
}

// ---------------- final FC GEMM: out = A @ W^T + bias -----------------------
// Block = 8 waves. Block tile = 128 rows x 64 cols; the 64x32 weight panel per
// k-step is staged into LDS with async-to-LDS loads (double buffered) and
// shared by all 8 waves (8x reduction of W traffic vs direct loads).
__global__ void __launch_bounds__(256)
gemm_wmma_bias(const _Float16* __restrict__ A, const _Float16* __restrict__ W,
               const float* __restrict__ bias, float* __restrict__ out,
               int M, int N, int K) {
  __shared__ _Float16 sB[2][64 * LDS_ROW];

  int tid  = threadIdx.x;
  int wv   = tid >> 5;
  int lane = tid & 31;
  int bm = blockIdx.x & 31;          // M/128 = 32 block rows
  int bn = blockIdx.x >> 5;          // N/64  = 500 block cols
  int m0 = (bm << 7) + (wv << 4);    // this wave's 16-row tile
  int n0 = bn << 6;                  // block's 64-col panel
  int n  = lane & 15;

  // async stage of one 64x32 weight panel: 256 threads x 16B
  int srow  = tid >> 2;              // 0..63
  int spart = tid & 3;               // 4 x 16B per 64B row
  unsigned ldsOff[2];
#pragma unroll
  for (int bf = 0; bf < 2; ++bf)
    ldsOff[bf] = (unsigned)(uintptr_t)&sB[bf][srow * LDS_ROW + spart * 8];
  const _Float16* gsrc = W + (size_t)(n0 + srow) * K + spart * 8;

  v8f acc[4];
#pragma unroll
  for (int jx = 0; jx < 4; ++jx) {
    float bv = bias ? bias[n0 + jx * 16 + n] : 0.f;
#pragma unroll
    for (int i = 0; i < 8; ++i) acc[jx][i] = bv;
  }

  int nk = K >> 5;                   // 16 k-steps of 32
  async_copy_b128(ldsOff[0], gsrc);  // prime buffer 0 (k0 = 0)
  for (int i = 0; i < nk; ++i) {
    wait_asynccnt0();
    __syncthreads();                 // staged panel visible to whole block
    if (i + 1 < nk)                  // overlap next stage with compute
      async_copy_b128(ldsOff[(i + 1) & 1], gsrc + ((i + 1) << 5));
    const _Float16* panel = sB[i & 1];
    v16h a = load_afrag(A, K, m0, i << 5);
#pragma unroll
    for (int jx = 0; jx < 4; ++jx) {
      v16h bmf = load_bfrag_lds(panel, jx * 16);
      acc[jx] = __builtin_amdgcn_wmma_f32_16x16x32_f16(
          false, a, false, bmf, (short)0, acc[jx], false, false);
    }
    __syncthreads();                 // all reads done before panel is reused
  }

  int m = ((lane >> 4) << 3);
#pragma unroll
  for (int jx = 0; jx < 4; ++jx) {
    int col = n0 + jx * 16 + n;
#pragma unroll
    for (int r = 0; r < 8; ++r)
      out[(size_t)(m0 + m + r) * N + col] = acc[jx][r];
  }
}

// ---------------------------------------------------------------------------
extern "C" void kernel_launch(void* const* d_in, const int* in_sizes, int n_in,
                              void* d_out, int out_size, void* d_ws, size_t ws_size,
                              hipStream_t stream) {
  const int*   input_seq  = (const int*)  d_in[0];
  const int*   output_seq = (const int*)  d_in[1];
  const float* embed      = (const float*)d_in[2];
  const float* enc_Wih    = (const float*)d_in[3];
  const float* enc_Whh    = (const float*)d_in[4];
  const float* enc_b      = (const float*)d_in[5];
  const float* dec_Wih    = (const float*)d_in[6];
  const float* dec_Whh    = (const float*)d_in[7];
  const float* dec_b      = (const float*)d_in[8];
  const float* fc_W       = (const float*)d_in[9];
  const float* fc_b       = (const float*)d_in[10];
  float*       logits     = (float*)d_out;

  // workspace carve-out (256B aligned)
  char* wp = (char*)d_ws;
  auto alloc = [&](size_t bytes) {
    char* p = wp; wp += (bytes + 255) & ~(size_t)255; return p;
  };
  _Float16* encW   = (_Float16*)alloc((size_t)G4H * KENC * 2);       // [2048, 768]
  _Float16* decW   = (_Float16*)alloc((size_t)G4H * KDEC * 2);       // [2048,1280]
  _Float16* fcW16  = (_Float16*)alloc((size_t)VOCAB * HDIM * 2);     // [32000,512]
  _Float16* encEmb = (_Float16*)alloc((size_t)B_SZ * S_SRC * EDIM * 2);
  _Float16* decEmb = (_Float16*)alloc((size_t)B_SZ * T_TGT * EDIM * 2);
  float*    encOut = (float*)   alloc((size_t)B_SZ * S_SRC * HDIM * 4);
  float*    h      = (float*)   alloc((size_t)B_SZ * HDIM * 4);
  float*    c      = (float*)   alloc((size_t)B_SZ * HDIM * 4);
  _Float16* h16    = (_Float16*)alloc((size_t)B_SZ * HDIM * 2);
  _Float16* xhE    = (_Float16*)alloc((size_t)B_SZ * KENC * 2);
  _Float16* xhD    = (_Float16*)alloc((size_t)B_SZ * KDEC * 2);
  float*    scores = (float*)   alloc((size_t)B_SZ * S_SRC * 4);
  _Float16* decHs  = (_Float16*)alloc((size_t)B_SZ * T_TGT * HDIM * 2);

  // ---- one-time f16 conversions / gathers ----
  {
    long n = (long)G4H * KENC;
    convert_concat_f16<<<(int)((n + 255) / 256), 256, 0, stream>>>(
        enc_Wih, EDIM, enc_Whh, HDIM, encW, n);
  }
  {
    long n = (long)G4H * KDEC;
    convert_concat_f16<<<(int)((n + 255) / 256), 256, 0, stream>>>(
        dec_Wih, EDIM + HDIM, dec_Whh, HDIM, decW, n);
  }
  {
    long n = (long)VOCAB * HDIM;
    convert_concat_f16<<<(int)((n + 255) / 256), 256, 0, stream>>>(
        fc_W, HDIM, nullptr, 0, fcW16, n);
  }
  {
    long n = (long)B_SZ * S_SRC * EDIM;
    gather_embed_f16<<<(int)((n + 255) / 256), 256, 0, stream>>>(
        input_seq, embed, encEmb, n);
  }
  {
    long n = (long)B_SZ * T_TGT * EDIM;
    gather_embed_f16<<<(int)((n + 255) / 256), 256, 0, stream>>>(
        output_seq, embed, decEmb, n);
  }
  zero_state<<<(B_SZ * HDIM) / 256, 256, 0, stream>>>(h, c, h16);

  // ---- encoder scan: 512 fused steps ----
  for (int t = 0; t < S_SRC; ++t) {
    pack_enc_xh<<<(B_SZ * KENC) / 256, 256, 0, stream>>>(encEmb, h16, xhE, t);
    lstm_step_fused<<<16, 128, 0, stream>>>(xhE, KENC, encW, enc_b,
                                            h, c, h16, encOut, nullptr,
                                            t, S_SRC);
  }

  // ---- decoder scan with attention: 128 steps ----
  for (int t = 0; t < T_TGT; ++t) {
    attn_scores<<<(B_SZ * S_SRC * 32) / 256, 256, 0, stream>>>(encOut, h, scores);
    attn_softmax_ctx_pack<<<B_SZ, S_SRC, 0, stream>>>(scores, encOut, decEmb,
                                                      h16, xhD, t);
    lstm_step_fused<<<16, 128, 0, stream>>>(xhD, KDEC, decW, dec_b,
                                            h, c, h16, nullptr, decHs,
                                            t, T_TGT);
  }

  // ---- final vocab projection: [4096,512] x [512,32000] + bias ----
  {
    int M = B_SZ * T_TGT, N = VOCAB, K = HDIM;
    int blocks = (M / 128) * (N / 64);   // 32 * 500 = 16000
    gemm_wmma_bias<<<blocks, 256, 0, stream>>>(decHs, fcW16, fc_b, logits,
                                               M, N, K);
  }
}
